// TransformerDecoder_88450556493973
// MI455X (gfx1250) — compile-verified, matrix operand order checked by reading, not run on hardware
//
#include <hip/hip_runtime.h>

typedef __attribute__((ext_vector_type(16))) __bf16 v16bf;
typedef __attribute__((ext_vector_type(8)))  __bf16 v8bf;
typedef __attribute__((ext_vector_type(8)))  float  v8f;

#define TT 2048
#define BB 2
#define CC 512
#define HH 8
#define HD 64
#define FF 2048
#define LL 4
#define MM (TT*BB)
#define SS 2048
#define C3 (3*CC)

// ---------------- device helpers ----------------

__device__ __forceinline__ v16bf make_frag(const __bf16* p0, const __bf16* p1) {
  v8bf lo = *(const v8bf*)p0;
  v8bf hi = *(const v8bf*)p1;
  v16bf f;
#pragma unroll
  for (int i = 0; i < 8; i++) { f[i] = lo[i]; f[i + 8] = hi[i]; }
  return f;
}

// monotonic float <-> uint mapping so unsigned atomicMin/Max order == float order
__device__ __forceinline__ unsigned f2key(float f) {
  unsigned b = __float_as_uint(f);
  return (b & 0x80000000u) ? ~b : (b | 0x80000000u);
}
__device__ __forceinline__ float key2f(unsigned k) {
  unsigned b = (k & 0x80000000u) ? (k & 0x7FFFFFFFu) : ~k;
  return __uint_as_float(b);
}

#define WMMA_BF16(A, Bf, Cacc) \
  __builtin_amdgcn_wmma_f32_16x16x32_bf16(false, (A), false, (Bf), (short)0, (Cacc), false, false)

// ---------------- kernels ----------------

// xf = x + sinusoidal positional embedding (padding_idx=0 semantics)
__global__ void add_pos_kernel(const float* __restrict__ x, const int* __restrict__ tokens,
                               float* __restrict__ xf) {
  int idx = blockIdx.x * blockDim.x + threadIdx.x;
  if (idx >= TT * BB * CC) return;
  int c = idx % CC;
  int b = (idx / CC) % BB;
  int t = idx / (CC * BB);
  int tok = tokens[b * TT + t];
  float pe = 0.f;
  if (tok != 0) {
    float pos = (float)(t + 1);
    int j = (c < CC / 2) ? c : (c - CC / 2);
    float freq = expf((float)j * (-logf(10000.f) / (float)(CC / 2 - 1)));
    float ang = pos * freq;
    pe = (c < CC / 2) ? sinf(ang) : cosf(ang);
  }
  xf[idx] = x[idx] + pe;
}

__global__ void f2bf_kernel(const float* __restrict__ in, __bf16* __restrict__ out, int n) {
  int i = blockIdx.x * blockDim.x + threadIdx.x;
  if (i < n) out[i] = (__bf16)in[i];
}

__global__ void init_mm_kernel(unsigned* mm) { mm[0] = 0xFFFFFFFFu; mm[1] = 0u; }

// Out[M,N] = A[M,K](bf16) @ W[N,K]^T(bf16) + bias ; optional ReLU.
// One wave = 32x64 output tile: 2 A-fragments x 4 B-fragments = 8 WMMAs per
// K-step of 32; B-fragments reused across both A-tiles (~21 flop/byte from L2).
__global__ __launch_bounds__(256)
void gemm_bf16_kernel(const __bf16* __restrict__ A, const __bf16* __restrict__ W,
                      const float* __restrict__ bias, float* __restrict__ Out,
                      int Mr, int N, int K, int relu) {
  int lane = threadIdx.x & 31;
  int wid = (blockIdx.x * blockDim.x + threadIdx.x) >> 5;
  int nt64 = N >> 6;
  int mt = wid / nt64;       // 32-row tile index
  int nt = wid % nt64;       // 64-col tile index
  if (mt >= (Mr >> 5)) return;
  int half = (lane < 16) ? 0 : 8;
  const __bf16* arow0 = A + (size_t)((mt << 5) + (lane & 15)) * K;
  const __bf16* arow1 = arow0 + (size_t)16 * K;
  v8f acc[2][4];
#pragma unroll
  for (int i = 0; i < 2; i++)
#pragma unroll
    for (int j = 0; j < 4; j++) acc[i][j] = (v8f){0, 0, 0, 0, 0, 0, 0, 0};
  for (int kb = 0; kb < K; kb += 32) {
    v16bf af0 = make_frag(arow0 + kb + half, arow0 + kb + half + 16);
    v16bf af1 = make_frag(arow1 + kb + half, arow1 + kb + half + 16);
    const __bf16* w0 = W + (size_t)((nt << 6) + (lane & 15)) * K + kb + half;
    v16bf b0 = make_frag(w0, w0 + 16);
    v16bf b1 = make_frag(w0 + (size_t)16 * K, w0 + (size_t)16 * K + 16);
    v16bf b2 = make_frag(w0 + (size_t)32 * K, w0 + (size_t)32 * K + 16);
    v16bf b3 = make_frag(w0 + (size_t)48 * K, w0 + (size_t)48 * K + 16);
    acc[0][0] = WMMA_BF16(af0, b0, acc[0][0]);
    acc[0][1] = WMMA_BF16(af0, b1, acc[0][1]);
    acc[0][2] = WMMA_BF16(af0, b2, acc[0][2]);
    acc[0][3] = WMMA_BF16(af0, b3, acc[0][3]);
    acc[1][0] = WMMA_BF16(af1, b0, acc[1][0]);
    acc[1][1] = WMMA_BF16(af1, b1, acc[1][1]);
    acc[1][2] = WMMA_BF16(af1, b2, acc[1][2]);
    acc[1][3] = WMMA_BF16(af1, b3, acc[1][3]);
  }
  int colb = (nt << 6) + (lane & 15);
#pragma unroll
  for (int i = 0; i < 2; i++) {
    int rowb = (mt << 5) + i * 16 + ((lane < 16) ? 0 : 8);
#pragma unroll
    for (int j = 0; j < 4; j++) {
      int col = colb + j * 16;
      float bv = bias[col];
#pragma unroll
      for (int r = 0; r < 8; r++) {
        float v = acc[i][j][r] + bv;
        if (relu) v = fmaxf(v, 0.f);
        Out[(size_t)(rowb + r) * N + col] = v;
      }
    }
  }
}

// Pass 1: scores = scale*(q k^T), masked -> 0, global min/max via ordered-uint atomics.
__global__ __launch_bounds__(256)
void attn_minmax_kernel(const __bf16* __restrict__ Q, int ldq,
                        const __bf16* __restrict__ Km, int ldk,
                        unsigned* __restrict__ mm, float scale, int masked) {
  int lane = threadIdx.x & 31;
  int wv = threadIdx.x >> 5;
  int ttile = blockIdx.x * (blockDim.x >> 5) + wv;
  int bh = blockIdx.y;
  int b = bh / HH, h = bh % HH;
  int half = (lane < 16) ? 0 : 8;
  int trow = (ttile << 4) + (lane & 15);
  const __bf16* qrow = Q + (size_t)(trow * BB + b) * ldq + h * HD;
  v16bf a0 = make_frag(qrow + half, qrow + half + 16);
  v16bf a1 = make_frag(qrow + 32 + half, qrow + 32 + half + 16);
  float lmin, lmax;
  if (masked) { lmin = 0.f; lmax = 0.f; }           // masked zeros participate in min/max
  else        { lmin = 3.4e38f; lmax = -3.4e38f; }
  int tmax = (ttile << 4) + 15;
  int nch = masked ? (tmax / 32 + 1) : (SS / 32);
  int rbase = (lane < 16) ? 0 : 8;
  for (int ch = 0; ch < nch; ch++) {
#pragma unroll
    for (int sub = 0; sub < 2; sub++) {
      int s0 = ch * 32 + sub * 16;
      const __bf16* krow = Km + (size_t)((s0 + (lane & 15)) * BB + b) * ldk + h * HD;
      v16bf b0 = make_frag(krow + half, krow + half + 16);
      v16bf b1 = make_frag(krow + 32 + half, krow + 32 + half + 16);
      v8f d = {0, 0, 0, 0, 0, 0, 0, 0};
      d = WMMA_BF16(a0, b0, d);
      d = WMMA_BF16(a1, b1, d);
      int scol = s0 + (lane & 15);
#pragma unroll
      for (int r = 0; r < 8; r++) {
        int t = (ttile << 4) + rbase + r;
        float v = d[r] * scale;
        if (masked && scol > t) v = 0.f;
        lmin = fminf(lmin, v);
        lmax = fmaxf(lmax, v);
      }
    }
  }
#pragma unroll
  for (int off = 16; off >= 1; off >>= 1) {
    lmin = fminf(lmin, __shfl_xor(lmin, off, 32));
    lmax = fmaxf(lmax, __shfl_xor(lmax, off, 32));
  }
  if (lane == 0) {
    atomicMin(&mm[0], f2key(lmin));
    atomicMax(&mm[1], f2key(lmax));
  }
}

// sumV[bh*HD+d] = sum_s v[s,d]
__global__ void sumv_kernel(const float* __restrict__ V, int ldv, float* __restrict__ sumV) {
  int idx = blockIdx.x * blockDim.x + threadIdx.x;
  if (idx >= BB * HH * HD) return;
  int d = idx % HD;
  int bh = idx / HD;
  int b = bh / HH, h = bh % HH;
  float s = 0.f;
  for (int t = 0; t < SS; t++) s += V[(size_t)(t * BB + b) * ldv + h * HD + d];
  sumV[idx] = s;
}

// vT[bh][d][s] = v_bf[s, b, h, d]
__global__ void vT_kernel(const __bf16* __restrict__ V, int ldv, __bf16* __restrict__ vT) {
  int idx = blockIdx.x * blockDim.x + threadIdx.x;
  if (idx >= BB * HH * HD * SS) return;
  int s = idx % SS;
  int rest = idx / SS;
  int d = rest % HD;
  int bh = rest / HD;
  int b = bh / HH, h = bh % HH;
  vT[idx] = V[(size_t)(s * BB + b) * ldv + h * HD + d];
}

// Pass 2 (fused): recompute score tiles via WMMA, LDS re-layout D->A fragments,
// WMMA against vT, epilogue applies global min-max norm via linearity:
//   a = (wm @ v - min * sumV) / (max - min)
// One wave per workgroup => __syncthreads is a legal per-wave LDS ordering point.
__global__ __launch_bounds__(32)
void attn_out_kernel(const __bf16* __restrict__ Q, int ldq,
                     const __bf16* __restrict__ Km, int ldk,
                     const __bf16* __restrict__ vT,
                     const float* __restrict__ sumV,
                     const unsigned* __restrict__ mm,
                     float* __restrict__ Aout, float scale, int masked) {
  __shared__ __bf16 sc[16][32];
  int lane = threadIdx.x & 31;
  int ttile = blockIdx.x;
  int bh = blockIdx.y;
  int b = bh / HH, h = bh % HH;
  int half = (lane < 16) ? 0 : 8;
  int rbase = half;
  int trow = (ttile << 4) + (lane & 15);
  const __bf16* qrow = Q + (size_t)(trow * BB + b) * ldq + h * HD;
  v16bf a0 = make_frag(qrow + half, qrow + half + 16);
  v16bf a1 = make_frag(qrow + 32 + half, qrow + 32 + half + 16);
  float minv = key2f(mm[0]);
  float maxv = key2f(mm[1]);
  float inv = 1.0f / (maxv - minv);
  v8f acc0 = {0}, acc1 = {0}, acc2 = {0}, acc3 = {0};
  int tmax = (ttile << 4) + 15;
  int nch = masked ? (tmax / 32 + 1) : (SS / 32);
  const __bf16* vbase = vT + (size_t)(bh * HD) * SS;
  for (int ch = 0; ch < nch; ch++) {
    int sc0 = ch * 32;
#pragma unroll
    for (int sub = 0; sub < 2; sub++) {
      int s0 = sc0 + sub * 16;
      const __bf16* krow = Km + (size_t)((s0 + (lane & 15)) * BB + b) * ldk + h * HD;
      v16bf b0 = make_frag(krow + half, krow + half + 16);
      v16bf b1 = make_frag(krow + 32 + half, krow + 32 + half + 16);
      v8f d = {0, 0, 0, 0, 0, 0, 0, 0};
      d = WMMA_BF16(a0, b0, d);
      d = WMMA_BF16(a1, b1, d);
      int scol = s0 + (lane & 15);
#pragma unroll
      for (int r = 0; r < 8; r++) {
        int t = (ttile << 4) + rbase + r;
        float v = d[r] * scale;
        if (masked && scol > t) v = 0.f;
        sc[rbase + r][sub * 16 + (lane & 15)] = (__bf16)v;
      }
    }
    __syncthreads();
    {
      const __bf16* srow = &sc[lane & 15][0];
      v16bf wf = make_frag(srow + half, srow + half + 16);
      const __bf16* v0 = vbase + (size_t)(lane & 15) * SS + sc0 + half;
      v16bf vb0 = make_frag(v0, v0 + 16);
      v16bf vb1 = make_frag(v0 + (size_t)16 * SS, v0 + (size_t)16 * SS + 16);
      v16bf vb2 = make_frag(v0 + (size_t)32 * SS, v0 + (size_t)32 * SS + 16);
      v16bf vb3 = make_frag(v0 + (size_t)48 * SS, v0 + (size_t)48 * SS + 16);
      acc0 = WMMA_BF16(wf, vb0, acc0);
      acc1 = WMMA_BF16(wf, vb1, acc1);
      acc2 = WMMA_BF16(wf, vb2, acc2);
      acc3 = WMMA_BF16(wf, vb3, acc3);
    }
    __syncthreads();
  }
  v8f accs[4] = {acc0, acc1, acc2, acc3};
#pragma unroll
  for (int j = 0; j < 4; j++) {
    int d = j * 16 + (lane & 15);
    float sv = sumV[bh * HD + d];
#pragma unroll
    for (int r = 0; r < 8; r++) {
      int t = (ttile << 4) + rbase + r;
      Aout[(size_t)(t * BB + b) * CC + h * HD + d] = (accs[j][r] - minv * sv) * inv;
    }
  }
}

// out = LayerNorm(res + y) * g + beta, one block per row (C=512, 256 thr x 2)
__global__ __launch_bounds__(256)
void add_ln_kernel(const float* __restrict__ res, const float* __restrict__ y,
                   const float* __restrict__ g, const float* __restrict__ beta,
                   float* __restrict__ out) {
  __shared__ float s1[256], s2[256];
  int row = blockIdx.x;
  int tid = threadIdx.x;
  const float* r0 = res + (size_t)row * CC;
  const float* y0 = y + (size_t)row * CC;
  float v0 = r0[tid] + y0[tid];
  float v1 = r0[tid + 256] + y0[tid + 256];
  s1[tid] = v0 + v1;
  s2[tid] = v0 * v0 + v1 * v1;
  __syncthreads();
  for (int off = 128; off >= 1; off >>= 1) {
    if (tid < off) { s1[tid] += s1[tid + off]; s2[tid] += s2[tid + off]; }
    __syncthreads();
  }
  float mu = s1[0] / (float)CC;
  float var = s2[0] / (float)CC - mu * mu;
  float rs = rsqrtf(var + 1e-20f);
  out[(size_t)row * CC + tid] = (v0 - mu) * rs * g[tid] + beta[tid];
  out[(size_t)row * CC + tid + 256] = (v1 - mu) * rs * g[tid + 256] + beta[tid + 256];
}

// ---------------- host orchestration ----------------

extern "C" void kernel_launch(void* const* d_in, const int* in_sizes, int n_in,
                              void* d_out, int out_size, void* d_ws, size_t ws_size,
                              hipStream_t stream) {
  (void)in_sizes; (void)n_in; (void)out_size; (void)ws_size;
  const float* x      = (const float*)d_in[0];
  const float* enc    = (const float*)d_in[1];
  const int*   tokens = (const int*)d_in[2];
  const float* sWqkv  = (const float*)d_in[3];
  const float* sbqkv  = (const float*)d_in[4];
  const float* sWo    = (const float*)d_in[5];
  const float* sbo    = (const float*)d_in[6];
  const float* cWqkv  = (const float*)d_in[7];
  const float* cbqkv  = (const float*)d_in[8];
  const float* cWo    = (const float*)d_in[9];
  const float* cbo    = (const float*)d_in[10];
  const float* f1w    = (const float*)d_in[11];
  const float* f1b    = (const float*)d_in[12];
  const float* f2w    = (const float*)d_in[13];
  const float* f2b    = (const float*)d_in[14];
  const float* ln1g   = (const float*)d_in[15];
  const float* ln1b   = (const float*)d_in[16];
  const float* ln2g   = (const float*)d_in[17];
  const float* ln2b   = (const float*)d_in[18];
  const float* ln3g   = (const float*)d_in[19];
  const float* ln3b   = (const float*)d_in[20];

  char* ws = (char*)d_ws;
  size_t off = 0;
  auto alloc = [&](size_t bytes) -> char* {
    char* p = ws + off;
    off = (off + bytes + 255) & ~(size_t)255;
    return p;
  };

  __bf16* wb_sqkv = (__bf16*)alloc((size_t)LL * C3 * CC * 2);
  __bf16* wb_swo  = (__bf16*)alloc((size_t)LL * CC * CC * 2);
  __bf16* wb_cqkv = (__bf16*)alloc((size_t)LL * C3 * CC * 2);
  __bf16* wb_cwo  = (__bf16*)alloc((size_t)LL * CC * CC * 2);
  __bf16* wb_f1   = (__bf16*)alloc((size_t)LL * FF * CC * 2);
  __bf16* wb_f2   = (__bf16*)alloc((size_t)LL * CC * FF * 2);
  float*  xf      = (float*)alloc((size_t)MM * CC * 4);
  __bf16* encbf   = (__bf16*)alloc((size_t)MM * CC * 2);
  __bf16* xbf     = (__bf16*)alloc((size_t)MM * CC * 2);
  float*  qkvf    = (float*)alloc((size_t)MM * C3 * 4);
  __bf16* qkvbf   = (__bf16*)alloc((size_t)MM * C3 * 2);
  float*  hf      = (float*)alloc((size_t)MM * FF * 4);
  __bf16* hbf     = (__bf16*)alloc((size_t)MM * FF * 2);
  float*  af      = (float*)alloc((size_t)MM * CC * 4);
  __bf16* abf     = (__bf16*)alloc((size_t)MM * CC * 2);
  float*  yf      = (float*)alloc((size_t)MM * CC * 4);
  __bf16* vT      = (__bf16*)alloc((size_t)BB * HH * HD * SS * 2);
  float*  sumv    = (float*)alloc((size_t)BB * HH * HD * 4);
  unsigned* mm    = (unsigned*)alloc(256);

  auto conv = [&](const float* src, __bf16* dst, int n) {
    f2bf_kernel<<<(n + 255) / 256, 256, 0, stream>>>(src, dst, n);
  };
  auto gemm = [&](const __bf16* A, const __bf16* W, const float* bias, float* Out,
                  int Mr, int N, int K, int relu) {
    int blocks = ((Mr / 32) * (N / 64)) / 8;
    gemm_bf16_kernel<<<blocks, 256, 0, stream>>>(A, W, bias, Out, Mr, N, K, relu);
  };

  // one-time: weight + encoder bf16 conversion, positional embedding
  conv(sWqkv, wb_sqkv, LL * C3 * CC);
  conv(sWo,   wb_swo,  LL * CC * CC);
  conv(cWqkv, wb_cqkv, LL * C3 * CC);
  conv(cWo,   wb_cwo,  LL * CC * CC);
  conv(f1w,   wb_f1,   LL * FF * CC);
  conv(f2w,   wb_f2,   LL * CC * FF);
  conv(enc,   encbf,   MM * CC);
  add_pos_kernel<<<(MM * CC) / 256, 256, 0, stream>>>(x, tokens, xf);

  const float scale = 0.125f;  // hd^-0.5, hd=64
  for (int l = 0; l < LL; l++) {
    // ---- self-attention ----
    conv(xf, xbf, MM * CC);
    gemm(xbf, wb_sqkv + (size_t)l * C3 * CC, sbqkv + l * C3, qkvf, MM, C3, CC, 0);
    conv(qkvf, qkvbf, MM * C3);
    init_mm_kernel<<<1, 1, 0, stream>>>(mm);
    attn_minmax_kernel<<<dim3(16, 16), 256, 0, stream>>>(qkvbf, C3, qkvbf + CC, C3, mm, scale, 1);
    sumv_kernel<<<4, 256, 0, stream>>>(qkvf + 2 * CC, C3, sumv);
    vT_kernel<<<(BB * HH * HD * SS) / 256, 256, 0, stream>>>(qkvbf + 2 * CC, C3, vT);
    attn_out_kernel<<<dim3(TT / 16, 16), 32, 0, stream>>>(qkvbf, C3, qkvbf + CC, C3,
                                                          vT, sumv, mm, af, scale, 1);
    conv(af, abf, MM * CC);
    gemm(abf, wb_swo + (size_t)l * CC * CC, sbo + l * CC, yf, MM, CC, CC, 0);
    add_ln_kernel<<<MM, 256, 0, stream>>>(xf, yf, ln1g + l * CC, ln1b + l * CC, xf);

    // ---- cross-attention (q from x, k/v from encoder) ----
    conv(xf, xbf, MM * CC);
    gemm(xbf, wb_cqkv + (size_t)l * C3 * CC, cbqkv + l * C3, qkvf, MM, CC, CC, 0);
    gemm(encbf, wb_cqkv + (size_t)l * C3 * CC + (size_t)CC * CC, cbqkv + l * C3 + CC,
         qkvf + (size_t)MM * CC, MM, 2 * CC, CC, 0);
    conv(qkvf, qkvbf, MM * C3);
    init_mm_kernel<<<1, 1, 0, stream>>>(mm);
    attn_minmax_kernel<<<dim3(16, 16), 256, 0, stream>>>(qkvbf, CC, qkvbf + (size_t)MM * CC,
                                                         2 * CC, mm, scale, 0);
    sumv_kernel<<<4, 256, 0, stream>>>(qkvf + (size_t)MM * CC + CC, 2 * CC, sumv);
    vT_kernel<<<(BB * HH * HD * SS) / 256, 256, 0, stream>>>(qkvbf + (size_t)MM * CC + CC,
                                                             2 * CC, vT);
    attn_out_kernel<<<dim3(TT / 16, 16), 32, 0, stream>>>(qkvbf, CC, qkvbf + (size_t)MM * CC,
                                                          2 * CC, vT, sumv, mm, af, scale, 0);
    conv(af, abf, MM * CC);
    gemm(abf, wb_cwo + (size_t)l * CC * CC, cbo + l * CC, yf, MM, CC, CC, 0);
    add_ln_kernel<<<MM, 256, 0, stream>>>(xf, yf, ln2g + l * CC, ln2b + l * CC, xf);

    // ---- FFN ----
    conv(xf, xbf, MM * CC);
    gemm(xbf, wb_f1 + (size_t)l * FF * CC, f1b + l * FF, hf, MM, FF, CC, 1);
    conv(hf, hbf, MM * FF);
    gemm(hbf, wb_f2 + (size_t)l * CC * FF, f2b + l * CC, yf, MM, CC, FF, 0);
    add_ln_kernel<<<MM, 256, 0, stream>>>(xf, yf, ln3g + l * CC, ln3b + l * CC, xf);
  }

  hipMemcpyAsync(d_out, xf, (size_t)MM * CC * sizeof(float),
                 hipMemcpyDeviceToDevice, stream);
}
